// PatchCorrespondenceAggregation_4647154614609
// MI455X (gfx1250) — matrix operand order repaired and checked by reading, not compiled, not run on hardware
//
#include <hip/hip_runtime.h>
#include <hip/hip_bf16.h>

typedef __attribute__((ext_vector_type(16))) __bf16 v16bf;
typedef __attribute__((ext_vector_type(8)))  float  v8f;

constexpr int Bn = 2, Cn = 64, Hn = 128, Wn = 128;
constexpr int PIX_TILES = Bn * Hn * (Wn / 16);   // 2048 tiles of 16 pixels
constexpr int PW1 = Wn + 2,  PH1 = Hn + 2;       // 1-pixel halo pitch (130)
constexpr int PW6 = Wn + 12, PH6 = Hn + 12;      // 6-pixel halo pitch (140)

// ---- f32 -> bf16 round-to-nearest-even ----
__device__ inline unsigned short f2bf(float f) {
  union { float f; unsigned u; } v; v.f = f;
  return (unsigned short)((v.u + 0x7FFFu + ((v.u >> 16) & 1u)) >> 16);
}

// =====================================================================
// Zero-fill (16B per thread). Buffers sized to multiples of 16 bytes.
// =====================================================================
__global__ void zero_k(uint4* __restrict__ p, long long n) {
  long long t = (long long)blockIdx.x * blockDim.x + threadIdx.x;
  if (t < n) p[t] = make_uint4(0u, 0u, 0u, 0u);
}

// =====================================================================
// Pack conv weights (OC,IC,3,3) f32 into per-lane WMMA A-fragments:
// layout [oc_tile][tap(9)][chunk(IC/32)][lane(32)][16] bf16, contiguous
// per lane => one 32B load per fragment.
// A-fragment k order used here (contiguous, same as B): lane l: m=l&15, kbase=(l<16?0:16)
//   element i (0..15) -> K = kbase + i
// =====================================================================
__global__ void pack_weights_k(const float* __restrict__ w,
                               unsigned short* __restrict__ out,
                               int OC, int IC, int tiles) {
  int chunks = IC >> 5;
  long long total = (long long)tiles * 9 * chunks * 512; // 32 lanes * 16 elems
  long long tid = (long long)blockIdx.x * blockDim.x + threadIdx.x;
  if (tid >= total) return;
  int i   = (int)(tid & 15);
  int l   = (int)((tid >> 4) & 31);
  int c   = (int)((tid >> 9) % chunks);
  int tap = (int)((tid / (512LL * chunks)) % 9);
  int t   = (int)(tid / (512LL * chunks * 9));
  int m = l & 15;
  int kb = (l < 16) ? 0 : 16;
  int K = kb + i;
  int ic = c * 32 + K;
  int oc = t * 16 + m;
  int kh = tap / 3, kw = tap % 3;
  float v = 0.0f;
  if (oc < OC) v = w[(((size_t)oc * IC + ic) * 3 + kh) * 3 + kw];
  out[tid] = f2bf(v);
}

// =====================================================================
// Per-pixel inverse l2 norms of nbr and ref (over C).
// =====================================================================
__global__ void norms_k(const float* __restrict__ nbr,
                        const float* __restrict__ ref,
                        float* __restrict__ norms /* (B*H*W, 2) */) {
  int p = blockIdx.x * blockDim.x + threadIdx.x;   // 0 .. B*H*W-1
  if (p >= Bn * Hn * Wn) return;
  int b = p / (Hn * Wn), pix = p % (Hn * Wn);
  float sn = 0.f, sr = 0.f;
  for (int c = 0; c < Cn; ++c) {
    float a = nbr[((size_t)b * Cn + c) * Hn * Wn + pix];
    float r = ref[((size_t)b * Cn + c) * Hn * Wn + pix];
    sn += a * a; sr += r * r;
  }
  float nn = sqrtf(sn), nr = sqrtf(sr);
  norms[(size_t)p * 2 + 0] = 1.0f / fmaxf(nn, 1e-12f);  // nbr
  norms[(size_t)p * 2 + 1] = 1.0f / fmaxf(nr, 1e-12f);  // ref
}

// =====================================================================
// Build NHWC activation buffers (interiors of halo-padded images):
//   X1  bf16 (B,130,130,128) = concat(nbr, ref)   -> conv1 input
//   XN  bf16 (B,140,140,64)  = nbr (6-px halo)    -> fused gather input
//   F1N f32  (B,H,W,64)      = l2norm(ref)        -> correlation
//   F2N f32  (B,H,W,64)      = l2norm(nbr)        -> correlation
//   X2  bf16 (B,130,130,128) channels 64..127 = ref -> final conv input
// =====================================================================
__global__ void pack_acts_k(const float* __restrict__ nbr,
                            const float* __restrict__ ref,
                            const float* __restrict__ norms,
                            unsigned short* __restrict__ X1,
                            unsigned short* __restrict__ XN,
                            float* __restrict__ F1N,
                            float* __restrict__ F2N,
                            unsigned short* __restrict__ X2) {
  int tid = blockIdx.x * blockDim.x + threadIdx.x;   // 64 * B*H*W threads
  const int NP = Bn * Hn * Wn;
  if (tid >= Cn * NP) return;
  int c = tid / NP, p = tid % NP;
  int b = p / (Hn * Wn), pix = p % (Hn * Wn);
  int h = pix / Wn, w = pix % Wn;
  float vn = nbr[((size_t)b * Cn + c) * Hn * Wn + pix];
  float vr = ref[((size_t)b * Cn + c) * Hn * Wn + pix];
  size_t pp1 = ((size_t)b * PH1 + (h + 1)) * PW1 + (w + 1);
  size_t pp6 = ((size_t)b * PH6 + (h + 6)) * PW6 + (w + 6);
  X1[pp1 * 128 + c]      = f2bf(vn);
  X1[pp1 * 128 + 64 + c] = f2bf(vr);
  XN[pp6 * 64 + c]       = f2bf(vn);
  F1N[(size_t)p * 64 + c] = vr * norms[(size_t)p * 2 + 1];
  F2N[(size_t)p * 64 + c] = vn * norms[(size_t)p * 2 + 0];
  X2[pp1 * 128 + 64 + c] = f2bf(vr);
}

// =====================================================================
// Generic implicit-GEMM 3x3 conv (pad 1) on halo-padded NHWC bf16 input
// (pitch 130x130; all taps read in-bounds -> branch-free inner loop).
// One wave per block; tile = 16 ocs x 16 pixels (one row segment).
// out_mode 0: bf16 NHWC into 1-px halo-padded image (stride out_ld, +coff)
// out_mode 1: f32 NCHW dense (channel stride = OC)
// =====================================================================
__global__ void conv3x3_wmma_k(const unsigned short* __restrict__ x,
                               const unsigned short* __restrict__ wp,
                               const float* __restrict__ bias,
                               void* __restrict__ outp,
                               int IC, int OC, int out_mode,
                               int out_ld, int out_coff) {
  int tile = blockIdx.x;                // pixel tile
  int t    = blockIdx.y;                // oc tile
  int b = tile / (Hn * (Wn / 16));
  int r = tile % (Hn * (Wn / 16));
  int h = r / (Wn / 16);
  int w0 = (r % (Wn / 16)) * 16;
  int l  = threadIdx.x;                 // 0..31
  int n  = l & 15;
  int hf = l >> 4;
  int w  = w0 + n;
  int chunks = IC >> 5;

  v8f acc = {};
#pragma unroll
  for (int kh = 0; kh < 3; ++kh) {
    // padded input coords: (h+kh, w+kw), always in [0,130)
    const unsigned short* xrow = x + ((size_t)(b * PH1 + (h + kh)) * PW1) * IC;
#pragma unroll
    for (int kw = 0; kw < 3; ++kw) {
      const unsigned short* xpix = xrow + (size_t)(w + kw) * IC + hf * 16;
      const unsigned short* wtap =
          wp + (((size_t)t * 9 + (kh * 3 + kw)) * chunks) * 512 + (size_t)l * 16;
      for (int c = 0; c < chunks; ++c) {
        v16bf a  = *(const v16bf*)(wtap + (size_t)c * 512);
        v16bf bf = *(const v16bf*)(xpix + c * 32);
        acc = __builtin_amdgcn_wmma_f32_16x16x32_bf16(
            false, a, false, bf, (short)0, acc, false, false);
      }
    }
  }
  // C/D layout: lane n=l&15 is column; vgpr rr holds row M = rr + 8*hf
#pragma unroll
  for (int rr = 0; rr < 8; ++rr) {
    int oc = t * 16 + rr + hf * 8;
    if (oc >= OC) continue;
    float v = acc[rr] + bias[oc];
    if (out_mode == 0) {
      size_t pp1 = ((size_t)b * PH1 + (h + 1)) * PW1 + (w + 1);
      ((unsigned short*)outp)[pp1 * out_ld + out_coff + oc] = f2bf(v);
    } else {
      ((float*)outp)[(((size_t)b * OC + oc) * Hn + h) * Wn + w] = v;
    }
  }
}

// =====================================================================
// Correlation stage 1: q[off,b,y,x] = sum_c f1n[y-1,x-1] * f2n[y-1+ph-5, x-1+pw-5]
// q grid is (H+2)x(W+2) = 130x130.
// =====================================================================
__global__ void corr_q_k(const float* __restrict__ f1n,
                         const float* __restrict__ f2n,
                         float* __restrict__ q) {
  const int QW = Wn + 2, QH = Hn + 2;
  long long total = 121LL * Bn * QH * QW;
  long long tid = (long long)blockIdx.x * blockDim.x + threadIdx.x;
  if (tid >= total) return;
  int x = (int)(tid % QW);
  long long t1 = tid / QW;
  int y = (int)(t1 % QH);
  long long t2 = t1 / QH;
  int b = (int)(t2 % Bn);
  int off = (int)(t2 / Bn);
  int ph = off / 11, pw = off % 11;
  int r1 = y - 1, c1 = x - 1;
  int r2 = r1 + ph - 5, c2 = c1 + pw - 5;
  float s = 0.f;
  if (r1 >= 0 && r1 < Hn && c1 >= 0 && c1 < Wn &&
      r2 >= 0 && r2 < Hn && c2 >= 0 && c2 < Wn) {
    const float* p1 = f1n + ((size_t)(b * Hn + r1) * Wn + c1) * 64;
    const float* p2 = f2n + ((size_t)(b * Hn + r2) * Wn + c2) * 64;
#pragma unroll 8
    for (int c = 0; c < Cn; ++c) s += p1[c] * p2[c];
  }
  q[tid] = s;
}

// =====================================================================
// Correlation stage 2: 3x3 box-sum over q + top-4 (desc, ties -> lower idx,
// matching lax.top_k) -> idx (B,H,W,4)
// =====================================================================
__global__ void topk_k(const float* __restrict__ q, int* __restrict__ idxout) {
  const int QW = Wn + 2, QH = Hn + 2;
  int p = blockIdx.x * blockDim.x + threadIdx.x;
  if (p >= Bn * Hn * Wn) return;
  int b = p / (Hn * Wn), pix = p % (Hn * Wn);
  int h = pix / Wn, w = pix % Wn;
  float b0 = -1e30f, b1 = -1e30f, b2 = -1e30f, b3 = -1e30f;
  int j0 = 0, j1 = 0, j2 = 0, j3 = 0;
  for (int off = 0; off < 121; ++off) {
    const float* qb = q + (((size_t)off * Bn + b) * QH + h) * QW + w;
    float s = 0.f;
#pragma unroll
    for (int ki = 0; ki < 3; ++ki)
#pragma unroll
      for (int kj = 0; kj < 3; ++kj) s += qb[ki * QW + kj];
    if (s > b0)      { b3=b2;j3=j2; b2=b1;j2=j1; b1=b0;j1=j0; b0=s;j0=off; }
    else if (s > b1) { b3=b2;j3=j2; b2=b1;j2=j1; b1=s;j1=off; }
    else if (s > b2) { b3=b2;j3=j2; b2=s;j2=off; }
    else if (s > b3) { b3=s;j3=off; }
  }
  idxout[(size_t)p * 4 + 0] = j0;
  idxout[(size_t)p * 4 + 1] = j1;
  idxout[(size_t)p * 4 + 2] = j2;
  idxout[(size_t)p * 4 + 3] = j3;
}

// =====================================================================
// Fused: patch gather + per-pixel NN conv (256ch 3x3 pad1 -> 64ch, 3x3 out)
//        + mask-weighted aggregation + LeakyReLU(0.1)
// writes agg as bf16 into X2 channels 0..63 (halo-padded NHWC, ld=128).
// Block = 128 threads (4 waves); wave t handles oc tile t (16 ocs).
// 9 f32 accumulators (output positions); K loop = 8 chunks of 32 over
// nbr*C = 256; B fragments read directly from the 6-px-halo XN at the
// displaced per-pixel patch coordinates -> always in-bounds, branch-free.
// =====================================================================
__global__ void nn_agg_wmma_k(const unsigned short* __restrict__ xn,
                              const unsigned short* __restrict__ wnp,
                              const float* __restrict__ nn_b,
                              const int* __restrict__ idx,
                              const float* __restrict__ mask,
                              unsigned short* __restrict__ x2) {
  int l = threadIdx.x & 31;
  int t = threadIdx.x >> 5;             // oc tile 0..3
  int tile = blockIdx.x;
  int b = tile / (Hn * (Wn / 16));
  int r = tile % (Hn * (Wn / 16));
  int h = r / (Wn / 16);
  int w0 = (r % (Wn / 16)) * 16;
  int n  = l & 15;
  int hf = l >> 4;
  int w  = w0 + n;

  // this lane's pixel's 4 neighbor displacements; in 6-px-halo coords the
  // patch element (y,x) of neighbor j lives at row h+dh[j]+y, col w+dw[j]+x
  int dh[4], dw[4];
  const int* ip = idx + ((size_t)(b * Hn + h) * Wn + w) * 4;
#pragma unroll
  for (int j = 0; j < 4; ++j) {
    int id = ip[j];
    dh[j] = id / 11;      // ph in 0..10 ; halo row = h + ph + y  (ph-6+6)
    dw[j] = id % 11;
  }

  v8f zero = {};
  v8f acc[9];
#pragma unroll
  for (int p = 0; p < 9; ++p) acc[p] = zero;

#pragma unroll
  for (int c = 0; c < 8; ++c) {                 // K chunks over nbr*C = 256
    int kstart = c * 32 + hf * 16;
    const int nb  = kstart >> 6;                // neighbor (const after unroll)
    const int ic0 = kstart & 63;                // 16 consecutive channels
#pragma unroll
    for (int y = 0; y < 3; ++y) {
      int yy = h + dh[nb] + y;                  // 0..139, always valid
#pragma unroll
      for (int x = 0; x < 3; ++x) {
        int xx = w + dw[nb] + x;
        v16bf bf = *(const v16bf*)(
            xn + ((size_t)(b * PH6 + yy) * PW6 + xx) * 64 + ic0);
        // taps (ki,kj) hitting patch coord (y,x): out pos i=y-ki+1, j=x-kj+1
#pragma unroll
        for (int ki = 0; ki < 3; ++ki) {
          int i = y - ki + 1;
          if (i < 0 || i > 2) continue;
#pragma unroll
          for (int kj = 0; kj < 3; ++kj) {
            int j = x - kj + 1;
            if (j < 0 || j > 2) continue;
            int pos = i * 3 + j;
            v16bf a = *(const v16bf*)(
                wnp + ((((size_t)t * 9 + (ki * 3 + kj)) * 8 + c) * 32 + l) * 16);
            acc[pos] = __builtin_amdgcn_wmma_f32_16x16x32_bf16(
                false, a, false, bf, (short)0, acc[pos], false, false);
          }
        }
      }
    }
  }

  // mask aggregation: agg[oc] = sum_pos (L[pos]+bias) * mask[g(oc),pos]
  const float* mb = mask + (size_t)b * 72 * Hn * Wn + (size_t)h * Wn + w;
#pragma unroll
  for (int rr = 0; rr < 8; ++rr) {
    int oc = t * 16 + rr + hf * 8;              // 0..63
    int g = oc >> 3;
    float bias = nn_b[oc];
    float s = 0.f;
#pragma unroll
    for (int pos = 0; pos < 9; ++pos) {
      float m = mb[(size_t)(g * 9 + pos) * Hn * Wn];
      s += (acc[pos][rr] + bias) * m;
    }
    s = (s >= 0.f) ? s : 0.1f * s;              // LeakyReLU(0.1)
    size_t pp1 = ((size_t)b * PH1 + (h + 1)) * PW1 + (w + 1);
    x2[pp1 * 128 + oc] = f2bf(s);
  }
}

// =====================================================================
extern "C" void kernel_launch(void* const* d_in, const int* in_sizes, int n_in,
                              void* d_out, int out_size, void* d_ws, size_t ws_size,
                              hipStream_t stream) {
  const float* nbr = (const float*)d_in[0];
  const float* ref = (const float*)d_in[1];
  const float* w1  = (const float*)d_in[2];  const float* b1 = (const float*)d_in[3];
  const float* w2  = (const float*)d_in[4];  const float* b2 = (const float*)d_in[5];
  const float* wm  = (const float*)d_in[6];  const float* bm = (const float*)d_in[7];
  const float* wn  = (const float*)d_in[8];  const float* bn = (const float*)d_in[9];
  const float* wa  = (const float*)d_in[10]; const float* ba = (const float*)d_in[11];
  float* out = (float*)d_out;

  // ---- workspace partition (all offsets 256B aligned) ----
  char* ws = (char*)d_ws;
  size_t off = 0;
  auto alloc = [&](size_t bytes) {
    size_t o = off;
    off = (off + bytes + 255) & ~(size_t)255;
    return o;
  };
  const size_t NP   = (size_t)Bn * Hn * Wn;
  const size_t NPP1 = (size_t)Bn * PH1 * PW1;     // 1-px halo pixels
  const size_t NPP6 = (size_t)Bn * PH6 * PW6;     // 6-px halo pixels
  unsigned short* X1  = (unsigned short*)(ws + alloc(NPP1 * 128 * 2));
  unsigned short* XN  = (unsigned short*)(ws + alloc(NPP6 * 64 * 2));
  unsigned short* Y1  = (unsigned short*)(ws + alloc(NPP1 * 64 * 2));
  unsigned short* Y2  = (unsigned short*)(ws + alloc(NPP1 * 64 * 2));
  unsigned short* X2  = (unsigned short*)(ws + alloc(NPP1 * 128 * 2));
  float* F1N   = (float*)(ws + alloc(NP * 64 * 4));
  float* F2N   = (float*)(ws + alloc(NP * 64 * 4));
  float* NORMS = (float*)(ws + alloc(NP * 2 * 4));
  float* Q     = (float*)(ws + alloc(121ULL * Bn * PH1 * PW1 * 4));
  float* MASK  = (float*)(ws + alloc((size_t)Bn * 72 * Hn * Wn * 4));
  int*   IDX   = (int*)  (ws + alloc(NP * 4 * 4));
  unsigned short* W1p = (unsigned short*)(ws + alloc(4ULL * 9 * 4 * 512 * 2));
  unsigned short* W2p = (unsigned short*)(ws + alloc(4ULL * 9 * 2 * 512 * 2));
  unsigned short* Wmp = (unsigned short*)(ws + alloc(5ULL * 9 * 2 * 512 * 2));
  unsigned short* Wnp = (unsigned short*)(ws + alloc(4ULL * 9 * 8 * 512 * 2));
  unsigned short* Wap = (unsigned short*)(ws + alloc(4ULL * 9 * 4 * 512 * 2));

  auto gsz = [](long long n, int b) { return (unsigned)((n + b - 1) / b); };

  // ---- zero halo-padded activation buffers (interiors overwritten below) ----
  auto zfill = [&](unsigned short* p, size_t ushorts) {
    long long n16 = (long long)(ushorts * 2 / 16);
    zero_k<<<gsz(n16, 256), 256, 0, stream>>>((uint4*)p, n16);
  };
  zfill(X1, NPP1 * 128);
  zfill(XN, NPP6 * 64);
  zfill(Y1, NPP1 * 64);
  zfill(Y2, NPP1 * 64);
  zfill(X2, NPP1 * 128);

  // ---- pack weights into WMMA A-fragment layout ----
  pack_weights_k<<<gsz(4LL*9*4*512, 256), 256, 0, stream>>>(w1, W1p, 64, 128, 4);
  pack_weights_k<<<gsz(4LL*9*2*512, 256), 256, 0, stream>>>(w2, W2p, 64, 64, 4);
  pack_weights_k<<<gsz(5LL*9*2*512, 256), 256, 0, stream>>>(wm, Wmp, 72, 64, 5);
  pack_weights_k<<<gsz(4LL*9*8*512, 256), 256, 0, stream>>>(wn, Wnp, 64, 256, 4);
  pack_weights_k<<<gsz(4LL*9*4*512, 256), 256, 0, stream>>>(wa, Wap, 64, 128, 4);

  // ---- activations: norms + NHWC/bf16 packing ----
  norms_k<<<gsz((long long)NP, 256), 256, 0, stream>>>(nbr, ref, NORMS);
  pack_acts_k<<<gsz((long long)NP * 64, 256), 256, 0, stream>>>(
      nbr, ref, NORMS, X1, XN, F1N, F2N, X2);

  // ---- mask branch: conv1 -> conv2 -> mask conv (all WMMA) ----
  conv3x3_wmma_k<<<dim3(PIX_TILES, 4), 32, 0, stream>>>(X1, W1p, b1, (void*)Y1,
                                                        128, 64, 0, 64, 0);
  conv3x3_wmma_k<<<dim3(PIX_TILES, 4), 32, 0, stream>>>(Y1, W2p, b2, (void*)Y2,
                                                        64, 64, 0, 64, 0);
  conv3x3_wmma_k<<<dim3(PIX_TILES, 5), 32, 0, stream>>>(Y2, Wmp, bm, (void*)MASK,
                                                        64, 72, 1, 0, 0);

  // ---- correlation + top-4 neighbor selection ----
  corr_q_k<<<gsz(121LL * Bn * PH1 * PW1, 256), 256, 0, stream>>>(F1N, F2N, Q);
  topk_k<<<gsz((long long)NP, 128), 128, 0, stream>>>(Q, IDX);

  // ---- fused gather + per-pixel NN conv + mask aggregation + LeakyReLU ----
  nn_agg_wmma_k<<<PIX_TILES, 128, 0, stream>>>(XN, Wnp, bn, IDX, MASK, X2);

  // ---- final conv (128 -> 64), f32 NCHW output ----
  conv3x3_wmma_k<<<dim3(PIX_TILES, 4), 32, 0, stream>>>(X2, Wap, ba, (void*)out,
                                                        128, 64, 1, 0, 0);
}
